// ElmanRNN_31722628448493
// MI455X (gfx1250) — matrix operand order repaired and checked by reading, not compiled, not run on hardware
//
#include <hip/hip_runtime.h>
#include <hip/hip_bf16.h>
#include <math.h>

typedef __attribute__((ext_vector_type(16))) _Float16 v16h;
typedef __attribute__((ext_vector_type(8)))  float    v8f;
typedef __attribute__((ext_vector_type(4)))  float    v4f;

#define B_  64
#define S_  512
#define I_  512
#define H_  1024

#define WMMA(acc, a, bb) \
    acc = __builtin_amdgcn_wmma_f32_16x16x32_f16(false, (a), false, (bb), (short)0, (acc), false, false)

// Build an A-fragment (16x32 f16, M = lane%16) from a row-major f32 row pointer
// already offset by +8*(lane>>4). Elements 0..7 <- K=[8g, 8g+8), 8..15 <- K=[16+8g, 16+8g+8).
__device__ __forceinline__ v16h make_afrag(const float* p) {
    v4f f0 = *(const v4f*)(p);
    v4f f1 = *(const v4f*)(p + 4);
    v4f f2 = *(const v4f*)(p + 16);
    v4f f3 = *(const v4f*)(p + 20);
    v16h a;
    a[0]=(_Float16)f0[0]; a[1]=(_Float16)f0[1]; a[2]=(_Float16)f0[2]; a[3]=(_Float16)f0[3];
    a[4]=(_Float16)f1[0]; a[5]=(_Float16)f1[1]; a[6]=(_Float16)f1[2]; a[7]=(_Float16)f1[3];
    a[8]=(_Float16)f2[0]; a[9]=(_Float16)f2[1]; a[10]=(_Float16)f2[2]; a[11]=(_Float16)f2[3];
    a[12]=(_Float16)f3[0]; a[13]=(_Float16)f3[1]; a[14]=(_Float16)f3[2]; a[15]=(_Float16)f3[3];
    return a;
}

// Pack f32 weight matrix W (H_ rows x Kdim cols, row-major) into f16 WMMA
// B-fragment tiles. Tile (nt, kt) -> 512 halves at ((nt*KT + kt)*512);
// within a tile: lane l, element e holds W[nt*16 + l%16, kt*32 + 8*(l>>4) + (e&7) + 16*(e>>3)],
// i.e. B-matrix element (K = 8*(l>>4) + (e&7) + 16*(e>>3), N = l%16) of the 32x16 tile of W^T.
__global__ void pack_w(const float* __restrict__ W, _Float16* __restrict__ P,
                       int Kdim, int KT) {
    int i = blockIdx.x * blockDim.x + threadIdx.x;
    int total = (H_ / 16) * KT * 512;
    if (i >= total) return;
    int tile = i >> 9;
    int idx  = i & 511;
    int l = idx >> 4;
    int e = idx & 15;
    int nt = tile / KT;
    int kt = tile - nt * KT;
    int n = nt * 16 + (l & 15);
    int k = kt * 32 + 8 * (l >> 4) + (e & 7) + 16 * (e >> 3);
    P[i] = (_Float16)W[(size_t)n * Kdim + k];
}

__global__ void init_misc(const float* __restrict__ b_ih, const float* __restrict__ b_hh,
                          float* __restrict__ bsum, float* __restrict__ zbuf) {
    int i = blockIdx.x * blockDim.x + threadIdx.x;
    if (i < H_) {
        bsum[i] = b_ih[i] + b_hh[i];
        zbuf[i] = 0.0f;
    }
}

// One recurrence timestep: h_t = tanh(x[:,t,:] @ Wih^T + bsum + h_{t-1} @ Whh^T),
// written to out[b, t, :]. h_{t-1} is read from hsrc + b*hstride (hstride=0 at t=0,
// else hsrc = out + (t-1)*H and hstride = S*H).
// Grid: 64 blocks x 32 threads (1 wave). Each wave: one 16-row M-tile x four 16-col N-tiles.
__global__ void __launch_bounds__(32)
rnn_step(const float* __restrict__ x_in,
         const _Float16* __restrict__ wih_p,
         const _Float16* __restrict__ whh_p,
         const float* __restrict__ bsum,
         const float* __restrict__ hsrc,
         size_t hstride,
         float* __restrict__ out,
         int t) {
    const int lane = threadIdx.x;         // 0..31
    const int task = blockIdx.x;          // 0..63
    const int mt   = task >> 4;           // 0..3   (batch tile)
    const int ntg  = task & 15;           // 0..15  (group of 4 hidden-col tiles)
    const int m = lane & 15;
    const int g = lane >> 4;
    const int b = mt * 16 + m;            // batch row this lane supplies for A
    const int nbase = ntg * 64;

    // Accumulators initialized with the per-column bias (C/D layout: N = lane%16).
    float b0 = bsum[nbase +  0 + m];
    float b1 = bsum[nbase + 16 + m];
    float b2 = bsum[nbase + 32 + m];
    float b3 = bsum[nbase + 48 + m];
    v8f c0 = {b0,b0,b0,b0,b0,b0,b0,b0};
    v8f c1 = {b1,b1,b1,b1,b1,b1,b1,b1};
    v8f c2 = {b2,b2,b2,b2,b2,b2,b2,b2};
    v8f c3 = {b3,b3,b3,b3,b3,b3,b3,b3};

    // ---- x_t @ W_ih^T : K = I_ = 512, 16 k-tiles of 32 ----
    {
        const float* arow = x_in + (size_t)b * (S_ * I_) + (size_t)t * I_ + 8 * g;
        const size_t ls = (size_t)lane * 16;
        for (int kt = 0; kt < I_ / 32; ++kt) {
            v16h a = make_afrag(arow + kt * 32);
            v16h w0 = *(const v16h*)(wih_p + ((size_t)(ntg * 4 + 0) * (I_/32) + kt) * 512 + ls);
            v16h w1 = *(const v16h*)(wih_p + ((size_t)(ntg * 4 + 1) * (I_/32) + kt) * 512 + ls);
            v16h w2 = *(const v16h*)(wih_p + ((size_t)(ntg * 4 + 2) * (I_/32) + kt) * 512 + ls);
            v16h w3 = *(const v16h*)(wih_p + ((size_t)(ntg * 4 + 3) * (I_/32) + kt) * 512 + ls);
            WMMA(c0, a, w0);
            WMMA(c1, a, w1);
            WMMA(c2, a, w2);
            WMMA(c3, a, w3);
        }
    }

    // ---- h_{t-1} @ W_hh^T : K = H_ = 1024, 32 k-tiles of 32 ----
    {
        const float* hrow = hsrc + (size_t)b * hstride + 8 * g;
        const size_t ls = (size_t)lane * 16;
        for (int kt = 0; kt < H_ / 32; ++kt) {
            v16h a = make_afrag(hrow + kt * 32);
            v16h w0 = *(const v16h*)(whh_p + ((size_t)(ntg * 4 + 0) * (H_/32) + kt) * 512 + ls);
            v16h w1 = *(const v16h*)(whh_p + ((size_t)(ntg * 4 + 1) * (H_/32) + kt) * 512 + ls);
            v16h w2 = *(const v16h*)(whh_p + ((size_t)(ntg * 4 + 2) * (H_/32) + kt) * 512 + ls);
            v16h w3 = *(const v16h*)(whh_p + ((size_t)(ntg * 4 + 3) * (H_/32) + kt) * 512 + ls);
            WMMA(c0, a, w0);
            WMMA(c1, a, w1);
            WMMA(c2, a, w2);
            WMMA(c3, a, w3);
        }
    }

    // ---- epilogue: tanh, store h_t into out[b, t, :] ----
    // C/D layout: VGPR r, lane l -> M = r + 8*(l>>4), N = l%16.
    #pragma unroll
    for (int r = 0; r < 8; ++r) {
        int br = mt * 16 + r + 8 * g;
        size_t base = (size_t)br * (S_ * H_) + (size_t)t * H_ + nbase + m;
        out[base +  0] = tanhf(c0[r]);
        out[base + 16] = tanhf(c1[r]);
        out[base + 32] = tanhf(c2[r]);
        out[base + 48] = tanhf(c3[r]);
    }
}

extern "C" void kernel_launch(void* const* d_in, const int* in_sizes, int n_in,
                              void* d_out, int out_size, void* d_ws, size_t ws_size,
                              hipStream_t stream) {
    const float* x_in = (const float*)d_in[0];   // (B, S, I)
    const float* W_ih = (const float*)d_in[1];   // (H, I)
    const float* W_hh = (const float*)d_in[2];   // (H, H)
    const float* b_ih = (const float*)d_in[3];   // (H)
    const float* b_hh = (const float*)d_in[4];   // (H)
    float* out = (float*)d_out;                  // (B, S, H)

    char* ws = (char*)d_ws;
    _Float16* wih_p = (_Float16*)(ws);                       // 1 MB packed f16 W_ih
    _Float16* whh_p = (_Float16*)(ws + (1u << 20));          // 2 MB packed f16 W_hh
    float*    bsum  = (float*)(ws + (3u << 20));             // 4 KB b_ih + b_hh
    float*    zbuf  = (float*)(ws + (3u << 20) + 4096);      // 4 KB zero row (h_{-1})

    // Pack weights into WMMA B-fragment order; compute bias sum; zero h_{-1} row.
    {
        int tot_ih = (H_ / 16) * (I_ / 32) * 512;            // 524288
        int tot_hh = (H_ / 16) * (H_ / 32) * 512;            // 1048576
        pack_w<<<(tot_ih + 255) / 256, 256, 0, stream>>>(W_ih, wih_p, I_, I_ / 32);
        pack_w<<<(tot_hh + 255) / 256, 256, 0, stream>>>(W_hh, whh_p, H_, H_ / 32);
        init_misc<<<(H_ + 255) / 256, 256, 0, stream>>>(b_ih, b_hh, bsum, zbuf);
    }

    // Sequential recurrence: one launch per timestep; step t reads h_{t-1}
    // straight from out[:, t-1, :] (stream order provides the dependency).
    for (int t = 0; t < S_; ++t) {
        const float* hsrc   = (t == 0) ? zbuf : (out + (size_t)(t - 1) * H_);
        size_t       hstride = (t == 0) ? 0 : (size_t)S_ * H_;
        rnn_step<<<64, 32, 0, stream>>>(x_in, wih_p, whh_p, bsum, hsrc, hstride, out, t);
    }
}
